// EncoderLayer_71476845740789
// MI455X (gfx1250) — matrix operand order repaired and checked by reading, not compiled, not run on hardware
//
#include <hip/hip_runtime.h>

// ---------------- types ----------------
typedef __bf16 bf16x16 __attribute__((ext_vector_type(16)));
typedef float  v8f     __attribute__((ext_vector_type(8)));

__device__ inline unsigned short f32_to_bf16(float f) {
    unsigned int u = __float_as_uint(f);
    unsigned int r = u + 0x7FFFu + ((u >> 16) & 1u);
    return (unsigned short)(r >> 16);
}
__device__ inline float bf16_to_f32(unsigned short h) {
    return __uint_as_float(((unsigned int)h) << 16);
}

__device__ inline v8f wmma_bf16(bf16x16 a, bf16x16 b, v8f c) {
    return __builtin_amdgcn_wmma_f32_16x16x32_bf16(false, a, false, b, (short)0, c,
                                                   false, false);
}

// ---- CDNA5 async global->LDS copy (ASYNCcnt path), 16B per lane ----
__device__ inline void async_copy_b128(const void* gptr, void* lptr) {
    unsigned       lo = (unsigned)(uintptr_t)lptr;             // LDS byte offset (low 32b of generic)
    unsigned long long ga = (unsigned long long)(uintptr_t)gptr;
    asm volatile("global_load_async_to_lds_b128 %0, %1, off"
                 :: "v"(lo), "v"(ga) : "memory");
}
__device__ inline void wait_async0() {
    asm volatile("s_wait_asynccnt 0x0" ::: "memory");
}

// ---------------- LayerNorm -> bf16 ----------------
__global__ __launch_bounds__(256) void ln_kernel(const float* __restrict__ x,
                                                 const float* __restrict__ g,
                                                 const float* __restrict__ b,
                                                 unsigned short* __restrict__ out) {
    __shared__ float red[256];
    int m = blockIdx.x;
    const float* xr = x + (size_t)m * 768;
    float s = 0.f;
    for (int i = threadIdx.x; i < 768; i += 256) s += xr[i];
    red[threadIdx.x] = s; __syncthreads();
    for (int o = 128; o > 0; o >>= 1) { if (threadIdx.x < o) red[threadIdx.x] += red[threadIdx.x + o]; __syncthreads(); }
    float mean = red[0] / 768.f;
    __syncthreads();
    float v = 0.f;
    for (int i = threadIdx.x; i < 768; i += 256) { float d = xr[i] - mean; v += d * d; }
    red[threadIdx.x] = v; __syncthreads();
    for (int o = 128; o > 0; o >>= 1) { if (threadIdx.x < o) red[threadIdx.x] += red[threadIdx.x + o]; __syncthreads(); }
    float inv = rsqrtf(red[0] / 768.f + 1e-5f);
    for (int i = threadIdx.x; i < 768; i += 256)
        out[(size_t)m * 768 + i] = f32_to_bf16((xr[i] - mean) * inv * g[i] + b[i]);
}

// ---------------- transpose + f32->bf16 : in[R][C] -> out[C][R] ----------------
__global__ __launch_bounds__(256) void tcvt_kernel(const float* __restrict__ in,
                                                   unsigned short* __restrict__ out,
                                                   int R, int C) {
    __shared__ float t[32][33];
    int tc = blockIdx.x * 32, tr = blockIdx.y * 32;
    int lx = threadIdx.x & 31, ly = threadIdx.x >> 5;
    for (int i = ly; i < 32; i += 8) {
        int r = tr + i, c = tc + lx;
        t[i][lx] = (r < R && c < C) ? in[(size_t)r * C + c] : 0.f;
    }
    __syncthreads();
    for (int i = ly; i < 32; i += 8) {
        int oc = tc + i, orr = tr + lx;
        if (oc < C && orr < R) out[(size_t)oc * R + orr] = f32_to_bf16(t[lx][i]);
    }
}

// ---------------- WMMA fragment loaders (LDS, row-major [rows][GLDT]) ----------------
#define GBM 128
#define GBN 64
#define GBK 64
#define GLDT 72

__device__ inline bf16x16 load_fragA(const unsigned short* lds, int row, int kbase, int lane) {
    const unsigned short* p = lds + row * GLDT;
    int half8 = (lane >> 4) << 3;
    union { unsigned int u[8]; bf16x16 v; } f;
#pragma unroll
    for (int r = 0; r < 8; r++) {
        int k = kbase + (half8 << 1) + (r << 1);
        f.u[r] = *(const unsigned int*)(p + k);
    }
    return f.v;
}
__device__ inline bf16x16 load_fragB(const unsigned short* lds, int row, int kbase, int lane) {
    const unsigned short* p = lds + row * GLDT;
    int half16 = (lane >> 4) << 4;
    union { unsigned int u[8]; bf16x16 v; } f;
#pragma unroll
    for (int r = 0; r < 8; r++) {
        int k = kbase + half16 + (r << 1);
        f.u[r] = *(const unsigned int*)(p + k);
    }
    return f.v;
}

// ---------------- dual-GEMM GLU kernel (async staged, double-buffered) ----------------
// out = f( silu(A@W1^T + b1) * (A@W2^T + b2) ); W stored transposed [N][K] bf16.
// epi: 0 = identity (f32), 1 = elu+1 (f32), 2 = (elu+1)*mask[m] (f32), 3 = identity (bf16)
__global__ __launch_bounds__(256) void glu_gemm_kernel(
    const unsigned short* __restrict__ A, const unsigned short* __restrict__ W1,
    const unsigned short* __restrict__ W2, const float* __restrict__ b1,
    const float* __restrict__ b2, const float* __restrict__ mask,
    float* __restrict__ outF, unsigned short* __restrict__ outH,
    int Kdim, int Ntot, int epi) {
    __shared__ unsigned short As[2][GBM * GLDT];
    __shared__ unsigned short W1s[2][GBN * GLDT];
    __shared__ unsigned short W2s[2][GBN * GLDT];
    int tid = threadIdx.x;
    int m0 = blockIdx.y * GBM, n0 = blockIdx.x * GBN;
    int wave = tid >> 5, lane = tid & 31;
    int wm = (wave & 3) * 32, wn = (wave >> 2) * 32;

    v8f zero = {0.f, 0.f, 0.f, 0.f, 0.f, 0.f, 0.f, 0.f};
    v8f acc1[2][2], acc2[2][2];
#pragma unroll
    for (int mt = 0; mt < 2; mt++)
#pragma unroll
        for (int nt = 0; nt < 2; nt++) { acc1[mt][nt] = zero; acc2[mt][nt] = zero; }

    int ar = tid >> 1, asg = (tid & 1) * 32;
    int wr = tid >> 2, wsg = (tid & 3) * 16;

    auto stage = [&](int buf, int k0) {
        const unsigned short* agp = A + (size_t)(m0 + ar) * Kdim + k0 + asg;
        unsigned short* alp = &As[buf][ar * GLDT + asg];
        async_copy_b128(agp,      alp);
        async_copy_b128(agp + 8,  alp + 8);
        async_copy_b128(agp + 16, alp + 16);
        async_copy_b128(agp + 24, alp + 24);
        const unsigned short* w1g = W1 + (size_t)(n0 + wr) * Kdim + k0 + wsg;
        async_copy_b128(w1g,     &W1s[buf][wr * GLDT + wsg]);
        async_copy_b128(w1g + 8, &W1s[buf][wr * GLDT + wsg + 8]);
        const unsigned short* w2g = W2 + (size_t)(n0 + wr) * Kdim + k0 + wsg;
        async_copy_b128(w2g,     &W2s[buf][wr * GLDT + wsg]);
        async_copy_b128(w2g + 8, &W2s[buf][wr * GLDT + wsg + 8]);
    };

    int nstage = Kdim / GBK;
    stage(0, 0);
    for (int s = 0; s < nstage; s++) {
        wait_async0();
        __syncthreads();                   // stage s visible to all waves; prev reads of other buf done
        if (s + 1 < nstage) stage((s + 1) & 1, (s + 1) * GBK);
        int buf = s & 1;
        const unsigned short* Ab  = As[buf];
        const unsigned short* W1b = W1s[buf];
        const unsigned short* W2b = W2s[buf];
#pragma unroll
        for (int kk = 0; kk < GBK; kk += 32) {
            bf16x16 af[2], wf1[2], wf2[2];
#pragma unroll
            for (int mt = 0; mt < 2; mt++) af[mt] = load_fragA(Ab, wm + mt * 16 + (lane & 15), kk, lane);
#pragma unroll
            for (int nt = 0; nt < 2; nt++) {
                wf1[nt] = load_fragB(W1b, wn + nt * 16 + (lane & 15), kk, lane);
                wf2[nt] = load_fragB(W2b, wn + nt * 16 + (lane & 15), kk, lane);
            }
#pragma unroll
            for (int mt = 0; mt < 2; mt++)
#pragma unroll
                for (int nt = 0; nt < 2; nt++) {
                    acc1[mt][nt] = wmma_bf16(af[mt], wf1[nt], acc1[mt][nt]);
                    acc2[mt][nt] = wmma_bf16(af[mt], wf2[nt], acc2[mt][nt]);
                }
        }
    }

    int colL = lane & 15, rh = (lane >> 4) * 8;
#pragma unroll
    for (int mt = 0; mt < 2; mt++)
#pragma unroll
        for (int nt = 0; nt < 2; nt++) {
            int n = n0 + wn + nt * 16 + colL;
            float bb1 = b1[n], bb2 = b2[n];
#pragma unroll
            for (int v = 0; v < 8; v++) {
                int m = m0 + wm + mt * 16 + rh + v;
                float x1 = acc1[mt][nt][v] + bb1;
                float x2 = acc2[mt][nt][v] + bb2;
                float g = (x1 / (1.f + __expf(-x1))) * x2;
                if (epi == 1)      g = (g > 0.f) ? g + 1.f : __expf(g);
                else if (epi == 2) g = ((g > 0.f) ? g + 1.f : __expf(g)) * mask[m];
                if (epi == 3) outH[(size_t)m * Ntot + n] = f32_to_bf16(g);
                else          outF[(size_t)m * Ntot + n] = g;
            }
        }
}

// ---------------- single-W GEMM (wo projection / MoE down), async double-buffered ----
// epi 0: out = acc + bias[n] + resid[m,n]              (writes)
// epi 1: out += (acc + bias[n]) * combine[m*4+expert]  (accumulates)
__global__ __launch_bounds__(256) void gemm_kernel(
    const unsigned short* __restrict__ A, const unsigned short* __restrict__ W,
    const float* __restrict__ bias, const float* __restrict__ resid,
    const float* __restrict__ combine, int expert,
    float* __restrict__ outF, int Kdim, int Ntot, int epi) {
    __shared__ unsigned short As[2][GBM * GLDT];
    __shared__ unsigned short Ws[2][GBN * GLDT];
    int tid = threadIdx.x;
    int m0 = blockIdx.y * GBM, n0 = blockIdx.x * GBN;
    int wave = tid >> 5, lane = tid & 31;
    int wm = (wave & 3) * 32, wn = (wave >> 2) * 32;

    v8f zero = {0.f, 0.f, 0.f, 0.f, 0.f, 0.f, 0.f, 0.f};
    v8f acc[2][2];
#pragma unroll
    for (int mt = 0; mt < 2; mt++)
#pragma unroll
        for (int nt = 0; nt < 2; nt++) acc[mt][nt] = zero;

    int ar = tid >> 1, asg = (tid & 1) * 32;
    int wr = tid >> 2, wsg = (tid & 3) * 16;

    auto stage = [&](int buf, int k0) {
        const unsigned short* agp = A + (size_t)(m0 + ar) * Kdim + k0 + asg;
        unsigned short* alp = &As[buf][ar * GLDT + asg];
        async_copy_b128(agp,      alp);
        async_copy_b128(agp + 8,  alp + 8);
        async_copy_b128(agp + 16, alp + 16);
        async_copy_b128(agp + 24, alp + 24);
        const unsigned short* wg = W + (size_t)(n0 + wr) * Kdim + k0 + wsg;
        async_copy_b128(wg,     &Ws[buf][wr * GLDT + wsg]);
        async_copy_b128(wg + 8, &Ws[buf][wr * GLDT + wsg + 8]);
    };

    int nstage = Kdim / GBK;
    stage(0, 0);
    for (int s = 0; s < nstage; s++) {
        wait_async0();
        __syncthreads();
        if (s + 1 < nstage) stage((s + 1) & 1, (s + 1) * GBK);
        int buf = s & 1;
        const unsigned short* Ab = As[buf];
        const unsigned short* Wb = Ws[buf];
#pragma unroll
        for (int kk = 0; kk < GBK; kk += 32) {
            bf16x16 af[2], wf[2];
#pragma unroll
            for (int mt = 0; mt < 2; mt++) af[mt] = load_fragA(Ab, wm + mt * 16 + (lane & 15), kk, lane);
#pragma unroll
            for (int nt = 0; nt < 2; nt++) wf[nt] = load_fragB(Wb, wn + nt * 16 + (lane & 15), kk, lane);
#pragma unroll
            for (int mt = 0; mt < 2; mt++)
#pragma unroll
                for (int nt = 0; nt < 2; nt++) acc[mt][nt] = wmma_bf16(af[mt], wf[nt], acc[mt][nt]);
        }
    }

    int colL = lane & 15, rh = (lane >> 4) * 8;
#pragma unroll
    for (int mt = 0; mt < 2; mt++)
#pragma unroll
        for (int nt = 0; nt < 2; nt++) {
            int n = n0 + wn + nt * 16 + colL;
            float bb = bias[n];
#pragma unroll
            for (int v = 0; v < 8; v++) {
                int m = m0 + wm + mt * 16 + rh + v;
                float a = acc[mt][nt][v] + bb;
                size_t oi = (size_t)m * Ntot + n;
                if (epi == 0) outF[oi] = a + resid[oi];
                else          outF[oi] += a * combine[m * 4 + expert];
            }
        }
}

// ------- kv partials: kvpart[c][bh][d][e] = sum_{s in chunk c} phik[s,d]*v[s,e] -------
__global__ __launch_bounds__(256) void kv_partial_kernel(const float* __restrict__ phik,
                                                         const float* __restrict__ vbuf,
                                                         float* __restrict__ kvpart,
                                                         float* __restrict__ kspart) {
    int bh = blockIdx.x, c = blockIdx.y;
    int b = bh / 12, h = bh % 12;
    __shared__ float pks[64 * 64];
    __shared__ float vs[64 * 64];
    int tid = threadIdx.x;
    int d = tid & 63, eg = tid >> 6;
    float acc[16];
#pragma unroll
    for (int j = 0; j < 16; j++) acc[j] = 0.f;
    float ka = 0.f;
    size_t base = ((size_t)b * 2048 + c * 256) * 768 + h * 64;
    for (int sc = 0; sc < 4; sc++) {
        for (int idx = tid; idx < 4096; idx += 256) {
            int s = idx >> 6, dd = idx & 63;
            size_t off = base + (size_t)(sc * 64 + s) * 768 + dd;
            pks[idx] = phik[off];
            vs[idx]  = vbuf[off];
        }
        __syncthreads();
        for (int s = 0; s < 64; s++) {
            float pk = pks[s * 64 + d];
            if (eg == 0) ka += pk;
            const float* vr = vs + s * 64 + eg * 16;
#pragma unroll
            for (int j = 0; j < 16; j++) acc[j] += pk * vr[j];
        }
        __syncthreads();
    }
    float* kvo = kvpart + ((size_t)c * 48 + bh) * 4096 + d * 64 + eg * 16;
#pragma unroll
    for (int j = 0; j < 16; j++) kvo[j] = acc[j];
    if (eg == 0) kspart[((size_t)c * 48 + bh) * 64 + d] = ka;
}

__global__ __launch_bounds__(256) void kv_reduce_kernel(const float* __restrict__ kvpart,
                                                        const float* __restrict__ kspart,
                                                        float* __restrict__ kv,
                                                        float* __restrict__ ksum) {
    int bh = blockIdx.x;
    for (int i = threadIdx.x; i < 4096; i += 256) {
        float s = 0.f;
        for (int c = 0; c < 8; c++) s += kvpart[((size_t)c * 48 + bh) * 4096 + i];
        kv[(size_t)bh * 4096 + i] = s;
    }
    if (threadIdx.x < 64) {
        float s = 0.f;
        for (int c = 0; c < 8; c++) s += kspart[((size_t)c * 48 + bh) * 64 + threadIdx.x];
        ksum[bh * 64 + threadIdx.x] = s;
    }
}

// ---------------- num/z + normalize, writes attn as bf16 ----------------
__global__ __launch_bounds__(256) void attn_kernel(const float* __restrict__ phiq,
                                                   const float* __restrict__ kv,
                                                   const float* __restrict__ ksum,
                                                   unsigned short* __restrict__ attnb) {
    int h = blockIdx.y, mc = blockIdx.x;
    int tid = threadIdx.x;
    __shared__ float kvs[64 * 64];
    __shared__ float kss[64];
    __shared__ float pq[4][64];
    int mbase = mc * 32;
    int b = mbase / 2048;
    int bh = b * 12 + h;
    for (int i = tid; i < 4096; i += 256) kvs[i] = kv[(size_t)bh * 4096 + i];
    if (tid < 64) kss[tid] = ksum[bh * 64 + tid];
    __syncthreads();
    int t4 = tid >> 6, e = tid & 63;
    for (int it = 0; it < 8; it++) {
        int m = mbase + it * 4 + t4;
        pq[t4][e] = phiq[(size_t)m * 768 + h * 64 + e];
        __syncthreads();
        float num = 0.f, z = 0.f;
        for (int d = 0; d < 64; d++) {
            float q = pq[t4][d];
            num += q * kvs[d * 64 + e];
            z += q * kss[d];
        }
        attnb[(size_t)m * 768 + h * 64 + e] = f32_to_bf16(num / (z + 1e-6f));
        __syncthreads();
    }
}

// ---------------- gate: softmax + top-2 combine weights ----------------
__global__ __launch_bounds__(256) void gate_kernel(const unsigned short* __restrict__ y,
                                                   const float* __restrict__ gw,
                                                   const float* __restrict__ gb,
                                                   float* __restrict__ combine) {
    int wave = threadIdx.x >> 5, lane = threadIdx.x & 31;
    int m = blockIdx.x * 8 + wave;
    float p0 = 0.f, p1 = 0.f, p2 = 0.f, p3 = 0.f;
    for (int i = lane; i < 768; i += 32) {
        float yv = bf16_to_f32(y[(size_t)m * 768 + i]);
        const float* g = gw + i * 4;
        p0 += yv * g[0]; p1 += yv * g[1]; p2 += yv * g[2]; p3 += yv * g[3];
    }
    for (int o = 16; o > 0; o >>= 1) {
        p0 += __shfl_down(p0, o, 32);
        p1 += __shfl_down(p1, o, 32);
        p2 += __shfl_down(p2, o, 32);
        p3 += __shfl_down(p3, o, 32);
    }
    if (lane == 0) {
        float l[4] = {p0 + gb[0], p1 + gb[1], p2 + gb[2], p3 + gb[3]};
        float mx = l[0];
        for (int e = 1; e < 4; e++) mx = fmaxf(mx, l[e]);
        float s = 0.f;
        for (int e = 0; e < 4; e++) { l[e] = __expf(l[e] - mx); s += l[e]; }
        for (int e = 0; e < 4; e++) l[e] /= s;
        int i1 = 0;
        for (int e = 1; e < 4; e++) if (l[e] > l[i1]) i1 = e;
        int i2 = (i1 == 0) ? 1 : 0;
        for (int e = 0; e < 4; e++) if (e != i1 && l[e] > l[i2]) i2 = e;
        float denom = l[i1] + l[i2] + 1e-6f;
        float c[4] = {0.f, 0.f, 0.f, 0.f};
        c[i1] = l[i1] / denom;
        c[i2] = l[i2] / denom;
        for (int e = 0; e < 4; e++) combine[m * 4 + e] = c[e];
    }
}

// ---------------- host launcher ----------------
extern "C" void kernel_launch(void* const* d_in, const int* in_sizes, int n_in,
                              void* d_out, int out_size, void* d_ws, size_t ws_size,
                              hipStream_t stream) {
    (void)in_sizes; (void)n_in; (void)out_size; (void)ws_size;
    const int Bq = 4, S = 2048, D = 768, F = 3072, E = 4;
    const int M = Bq * S; // 8192

    const float* x    = (const float*)d_in[0];
    const float* mask = (const float*)d_in[1];
    const float* qw1 = (const float*)d_in[2];  const float* qb1 = (const float*)d_in[3];
    const float* qw2 = (const float*)d_in[4];  const float* qb2 = (const float*)d_in[5];
    const float* kw1 = (const float*)d_in[6];  const float* kb1 = (const float*)d_in[7];
    const float* kw2 = (const float*)d_in[8];  const float* kb2 = (const float*)d_in[9];
    const float* vw1 = (const float*)d_in[10]; const float* vb1 = (const float*)d_in[11];
    const float* vw2 = (const float*)d_in[12]; const float* vb2 = (const float*)d_in[13];
    const float* wo  = (const float*)d_in[14]; const float* bo  = (const float*)d_in[15];
    const float* ew1 = (const float*)d_in[16]; const float* eb1 = (const float*)d_in[17];
    const float* ew3 = (const float*)d_in[18]; const float* eb3 = (const float*)d_in[19];
    const float* ew2 = (const float*)d_in[20]; const float* eb2 = (const float*)d_in[21];
    const float* gw  = (const float*)d_in[22]; const float* gb  = (const float*)d_in[23];
    const float* g1  = (const float*)d_in[24]; const float* be1 = (const float*)d_in[25];
    const float* g2  = (const float*)d_in[26]; const float* be2 = (const float*)d_in[27];

    size_t off = 0;
    auto take = [&](size_t bytes) -> char* {
        char* p = (char*)d_ws + off;
        off += (bytes + 255) & ~(size_t)255;
        return p;
    };
    unsigned short* x2   = (unsigned short*)take((size_t)M * D * 2);
    unsigned short* wt   = (unsigned short*)take((size_t)7 * D * D * 2);
    unsigned short* ew1t = (unsigned short*)take((size_t)E * F * D * 2);
    unsigned short* ew3t = (unsigned short*)take((size_t)E * F * D * 2);
    unsigned short* ew2t = (unsigned short*)take((size_t)E * D * F * 2);
    float* phiq  = (float*)take((size_t)M * D * 4);
    float* phik  = (float*)take((size_t)M * D * 4);
    float* vbuf  = (float*)take((size_t)M * D * 4);
    float* kvpart = (float*)take((size_t)8 * 48 * 4096 * 4);
    float* kspart = (float*)take((size_t)8 * 48 * 64 * 4);
    float* kvb   = (float*)take((size_t)48 * 64 * 64 * 4);
    float* ksum  = (float*)take((size_t)48 * 64 * 4);
    unsigned short* attnb = (unsigned short*)take((size_t)M * D * 2);
    unsigned short* ybf   = (unsigned short*)take((size_t)M * D * 2);
    float* comb  = (float*)take((size_t)M * E * 4);
    unsigned short* hbuf = (unsigned short*)take((size_t)M * F * 2);
    float* outF = (float*)d_out;

    // 1) LN1 -> x2 (bf16)
    ln_kernel<<<M, 256, 0, stream>>>(x, g1, be1, x2);

    // 2) weight transpose + bf16 convert
    dim3 tg(D / 32, D / 32);
    const float* wsrc[7] = {qw1, qw2, kw1, kw2, vw1, vw2, wo};
    for (int i = 0; i < 7; i++)
        tcvt_kernel<<<tg, 256, 0, stream>>>(wsrc[i], wt + (size_t)i * D * D, D, D);
    for (int e = 0; e < E; e++) {
        tcvt_kernel<<<dim3(F / 32, D / 32), 256, 0, stream>>>(ew1 + (size_t)e * D * F, ew1t + (size_t)e * F * D, D, F);
        tcvt_kernel<<<dim3(F / 32, D / 32), 256, 0, stream>>>(ew3 + (size_t)e * D * F, ew3t + (size_t)e * F * D, D, F);
        tcvt_kernel<<<dim3(D / 32, F / 32), 256, 0, stream>>>(ew2 + (size_t)e * F * D, ew2t + (size_t)e * D * F, F, D);
    }

    // 3) q/k/v GLU projections (WMMA)
    dim3 gqkv(D / GBN, M / GBM); // (12, 64)
    glu_gemm_kernel<<<gqkv, 256, 0, stream>>>(x2, wt + 0 * (size_t)D * D, wt + 1 * (size_t)D * D,
                                              qb1, qb2, mask, phiq, nullptr, D, D, 1);
    glu_gemm_kernel<<<gqkv, 256, 0, stream>>>(x2, wt + 2 * (size_t)D * D, wt + 3 * (size_t)D * D,
                                              kb1, kb2, mask, phik, nullptr, D, D, 2);
    glu_gemm_kernel<<<gqkv, 256, 0, stream>>>(x2, wt + 4 * (size_t)D * D, wt + 5 * (size_t)D * D,
                                              vb1, vb2, mask, vbuf, nullptr, D, D, 0);

    // 4) linear attention reductions (deterministic two-pass)
    kv_partial_kernel<<<dim3(48, 8), 256, 0, stream>>>(phik, vbuf, kvpart, kspart);
    kv_reduce_kernel<<<48, 256, 0, stream>>>(kvpart, kspart, kvb, ksum);
    attn_kernel<<<dim3(M / 32, 12), 256, 0, stream>>>(phiq, kvb, ksum, attnb);

    // 5) output projection + residual -> d_out
    gemm_kernel<<<dim3(D / GBN, M / GBM), 256, 0, stream>>>(attnb, wt + 6 * (size_t)D * D,
                                                            bo, x, nullptr, 0, outF, D, D, 0);

    // 6) LN2 -> ybf, gate
    ln_kernel<<<M, 256, 0, stream>>>(outF, g2, be2, ybf);
    gate_kernel<<<M / 8, 256, 0, stream>>>(ybf, gw, gb, comb);

    // 7) MoE: per expert GLU up + weighted down-projection accumulate into d_out
    for (int e = 0; e < E; e++) {
        glu_gemm_kernel<<<dim3(F / GBN, M / GBM), 256, 0, stream>>>(
            ybf, ew1t + (size_t)e * F * D, ew3t + (size_t)e * F * D,
            eb1 + (size_t)e * F, eb3 + (size_t)e * F, nullptr,
            nullptr, hbuf, D, F, 3);
        gemm_kernel<<<dim3(D / GBN, M / GBM), 256, 0, stream>>>(
            hbuf, ew2t + (size_t)e * D * F, eb2 + (size_t)e * D,
            nullptr, comb, e, outF, F, D, 1);
    }
}
